// Meta_Encoder_48490180772546
// MI455X (gfx1250) — matrix operand order repaired and checked by reading, not compiled, not run on hardware
//
#include <hip/hip_runtime.h>

#define N_NODES 100000
#define N_EDGES 1600000
#define NFEAT 256
#define NHID 128
#define MT_PAD 6252  // ceil(100000/16)=6250 padded up to a multiple of 4 row-tiles

typedef __bf16 v16bf __attribute__((ext_vector_type(16)));
typedef float  v8f   __attribute__((ext_vector_type(8)));

// ---------------- fragment-layout helpers (wave32 WMMA, 16-bit data) -------
// A 16x32 tile: lanes 0-15 hold K=0..7 & 16..23, lanes 16-31 hold K=8..15 & 24..31
__device__ __forceinline__ size_t a_index(int m, int k, int ktiles) {
  int mt = m >> 4, kt = k >> 5, kk = k & 31;
  int lane = (m & 15) + (((kk >> 3) & 1) << 4);
  int slot = ((kk >> 4) << 3) + (kk & 7);
  return (((size_t)(mt * ktiles + kt)) << 9) + (lane << 4) + slot;
}
// B 32x16 tile: lanes 0-15 hold K=0..7 & 16..23, lanes 16-31 hold K=8..15 & 24..31 (8 col-tiles for NHID=128)
__device__ __forceinline__ size_t b_index(int k, int n) {
  int kt = k >> 5, kk = k & 31, ct = n >> 4;
  int lane = (n & 15) + (((kk >> 3) & 1) << 4);
  int slot = ((kk >> 4) << 3) + (kk & 7);
  return (((size_t)(kt * 8 + ct)) << 9) + (lane << 4) + slot;
}

// ---------------- weight fp32 -> bf16 B-fragments --------------------------
__global__ __launch_bounds__(256) void prep_w_kernel(const float* __restrict__ W,
                                                     __bf16* __restrict__ Bf, int K) {
  int t = blockIdx.x * blockDim.x + threadIdx.x;
  if (t >= K * NHID) return;
  int k = t / NHID, n = t % NHID;
  Bf[b_index(k, n)] = (__bf16)W[t];
}

// ---------------- x fp32 -> bf16 A-fragments (K = 256), zero-pads tail -----
// One thread per (even-k, k+1) pair -> single packed 32-bit store.
__global__ __launch_bounds__(256) void cvt_x_kernel(const float* __restrict__ X,
                                                    __bf16* __restrict__ Af,
                                                    int n_rows, int K) {
  int ktiles = K >> 5;
  size_t p = (size_t)blockIdx.x * blockDim.x + threadIdx.x;
  size_t total = (size_t)MT_PAD * ktiles * 256;  // pairs
  if (p >= total) return;
  size_t frag = p >> 8;
  int within = (int)(p & 255);
  int lane = within >> 3;
  int slot = (within & 7) << 1;  // even slot
  int mt = (int)(frag / ktiles), kt = (int)(frag % ktiles);
  int kk = ((slot >> 3) << 4) + ((lane >> 4) << 3) + (slot & 7);
  int m = (mt << 4) + (lane & 15);
  int k = (kt << 5) + kk;
  float v0 = 0.0f, v1 = 0.0f;
  if (m < n_rows) {
    v0 = X[(size_t)m * K + k];
    v1 = X[(size_t)m * K + k + 1];
  }
  union { __bf16 b[2]; unsigned u; } pk;
  pk.b[0] = (__bf16)v0;
  pk.b[1] = (__bf16)v1;
  *(unsigned*)(Af + ((frag << 9) + (lane << 4) + slot)) = pk.u;
}

// ---------------- dense transform: support = A @ W via v_wmma bf16 ---------
// block = 256 threads = 8 waves; block covers 64 rows x 128 cols.
// A fragments (contiguous 4*ktiles*1KB region) are staged into LDS once via
// async global->LDS copies (ASYNCcnt), then read with ds_load per K-step.
// B comes straight from global (distinct col-tile per wave, W is L2-hot).
__global__ __launch_bounds__(256) void gemm_wmma_kernel(const v16bf* __restrict__ Af,
                                                        const v16bf* __restrict__ Bf,
                                                        float* __restrict__ out,
                                                        int n_rows, int ktiles) {
  __shared__ __align__(32) __bf16 sA[4 * 8 * 512];  // up to 32 KB (ktiles<=8)

  const int lane = threadIdx.x & 31;
  const int wave = threadIdx.x >> 5;  // col-tile 0..7
  const int mt0 = blockIdx.x * 4;

  // ---- async stage A block-tile: contiguous copy of 4*ktiles KB ----
  {
    const unsigned lds_base = (unsigned)(uintptr_t)(void*)sA;  // low 32 bits = LDS offset
    const char* gsrc = (const char*)Af + ((size_t)mt0 * ktiles << 10);
    const int nbytes = (4 * ktiles) << 10;
    for (int off = threadIdx.x * 16; off < nbytes; off += 256 * 16) {
      unsigned laddr = lds_base + off;
      unsigned long long gaddr = (unsigned long long)(gsrc + off);
      asm volatile("global_load_async_to_lds_b128 %0, %1, off"
                   :: "v"(laddr), "v"(gaddr) : "memory");
    }
    asm volatile("s_wait_asynccnt 0x0" ::: "memory");
    __syncthreads();
  }

  v8f acc[4];
  acc[0] = (v8f){}; acc[1] = (v8f){}; acc[2] = (v8f){}; acc[3] = (v8f){};

  for (int kt = 0; kt < ktiles; ++kt) {
    v16bf b = Bf[((size_t)(kt * 8 + wave) << 5) + lane];
    const v16bf* sfrag = (const v16bf*)sA;
    v16bf a0 = sfrag[((0 * ktiles + kt) << 5) + lane];
    v16bf a1 = sfrag[((1 * ktiles + kt) << 5) + lane];
    v16bf a2 = sfrag[((2 * ktiles + kt) << 5) + lane];
    v16bf a3 = sfrag[((3 * ktiles + kt) << 5) + lane];
    acc[0] = __builtin_amdgcn_wmma_f32_16x16x32_bf16(false, a0, false, b, (short)0, acc[0], false, false);
    acc[1] = __builtin_amdgcn_wmma_f32_16x16x32_bf16(false, a1, false, b, (short)0, acc[1], false, false);
    acc[2] = __builtin_amdgcn_wmma_f32_16x16x32_bf16(false, a2, false, b, (short)0, acc[2], false, false);
    acc[3] = __builtin_amdgcn_wmma_f32_16x16x32_bf16(false, a3, false, b, (short)0, acc[3], false, false);
  }

  // C/D layout: VGPR r -> M = r + 8*(lane>=16), N = lane%16
  const int ncol = wave * 16 + (lane & 15);
  const int mh = (lane >> 4) * 8;
#pragma unroll
  for (int mt = 0; mt < 4; ++mt) {
    int rowbase = (mt0 + mt) * 16 + mh;
#pragma unroll
    for (int r = 0; r < 8; ++r) {
      int row = rowbase + r;
      if (row < n_rows) out[(size_t)row * NHID + ncol] = acc[mt][r];
    }
  }
}

// ---------------- zero the aggregation buffer ------------------------------
__global__ __launch_bounds__(256) void zero_kernel(float4* __restrict__ p, size_t n4) {
  size_t i = (size_t)blockIdx.x * blockDim.x + threadIdx.x;
  if (i < n4) p[i] = make_float4(0.f, 0.f, 0.f, 0.f);
}

// ---------------- SpMM scatter: agg[row] += val * support[col] -------------
// thread t: edge e = t/32, feature quad q = t%32 (4 floats), L2-resident atomics
__global__ __launch_bounds__(256) void scatter_kernel(const int* __restrict__ erow,
                                                      const int* __restrict__ ecol,
                                                      const float* __restrict__ eval,
                                                      const float* __restrict__ support,
                                                      float* __restrict__ agg) {
  size_t t = (size_t)blockIdx.x * blockDim.x + threadIdx.x;
  size_t e = t >> 5;
  int q = (int)(t & 31);
  if (e >= (size_t)N_EDGES) return;
  int c = ecol[e];
  int r = erow[e];
  float v = eval[e];
  float4 s = ((const float4*)(support + (size_t)c * NHID))[q];
  float* dst = agg + (size_t)r * NHID + q * 4;
  atomicAdd(dst + 0, v * s.x);
  atomicAdd(dst + 1, v * s.y);
  atomicAdd(dst + 2, v * s.z);
  atomicAdd(dst + 3, v * s.w);
}

// ---------------- bias + ReLU; emit fp32 h (d_out) and bf16 A-frags --------
// One thread per bf16 pair: float2 store to out, packed u32 store to Af.
__global__ __launch_bounds__(256) void finalize_kernel(const float* __restrict__ agg,
                                                       const float* __restrict__ bias,
                                                       float* __restrict__ out,
                                                       __bf16* __restrict__ Af) {
  size_t t = (size_t)blockIdx.x * blockDim.x + threadIdx.x;
  int m = (int)(t >> 6);
  int k0 = (int)(t & 63) << 1;
  if (m >= MT_PAD * 16) return;
  float v0 = 0.0f, v1 = 0.0f;
  if (m < N_NODES) {
    const float* ap = agg + (size_t)m * NHID + k0;
    v0 = ap[0] + bias[k0];
    v1 = ap[1] + bias[k0 + 1];
    v0 = v0 > 0.0f ? v0 : 0.0f;
    v1 = v1 > 0.0f ? v1 : 0.0f;
    *(float2*)(out + (size_t)m * NHID + k0) = make_float2(v0, v1);
  }
  union { __bf16 b[2]; unsigned u; } pk;
  pk.b[0] = (__bf16)v0;
  pk.b[1] = (__bf16)v1;
  *(unsigned*)(Af + a_index(m, k0, NHID >> 5)) = pk.u;
}

extern "C" void kernel_launch(void* const* d_in, const int* in_sizes, int n_in,
                              void* d_out, int out_size, void* d_ws, size_t ws_size,
                              hipStream_t stream) {
  const float* x    = (const float*)d_in[0];
  const int*   erow = (const int*)d_in[1];
  const int*   ecol = (const int*)d_in[2];
  const float* eval = (const float*)d_in[3];
  const float* W1   = (const float*)d_in[4];
  const float* b1   = (const float*)d_in[5];
  const float* W2   = (const float*)d_in[6];
  const float* b2   = (const float*)d_in[7];
  float* out = (float*)d_out;

  char* ws = (char*)d_ws;
  const size_t szA = (size_t)MT_PAD * 8 * 512 * sizeof(__bf16);  // 51.2 MB (K=256 layout)
  const size_t szS = (size_t)N_NODES * NHID * sizeof(float);     // 51.2 MB
  size_t offA  = 0;
  size_t offS  = (offA + szA + 255) & ~(size_t)255;
  size_t offG  = (offS + szS + 255) & ~(size_t)255;
  size_t offW1 = (offG + szS + 255) & ~(size_t)255;
  size_t offW2 = offW1 + (size_t)NFEAT * NHID * sizeof(__bf16);

  __bf16* Af      = (__bf16*)(ws + offA);
  float*  support = (float*)(ws + offS);
  float*  agg     = (float*)(ws + offG);
  __bf16* Wf1     = (__bf16*)(ws + offW1);
  __bf16* Wf2     = (__bf16*)(ws + offW2);

  // Stage weights as bf16 B-fragments, x as bf16 A-fragments.
  prep_w_kernel<<<(NFEAT * NHID + 255) / 256, 256, 0, stream>>>(W1, Wf1, NFEAT);
  prep_w_kernel<<<(NHID * NHID + 255) / 256, 256, 0, stream>>>(W2, Wf2, NHID);
  {
    size_t pairs = (size_t)MT_PAD * (NFEAT >> 5) * 256;
    cvt_x_kernel<<<(unsigned)((pairs + 255) / 256), 256, 0, stream>>>(x, Af, N_NODES, NFEAT);
  }

  const size_t n4 = (size_t)N_NODES * NHID / 4;
  for (int l = 0; l < 5; ++l) {
    const int K = (l == 0) ? NFEAT : NHID;
    const __bf16* Wf = (l == 0) ? Wf1 : Wf2;
    const float* bias = (l == 0) ? b1 : b2;

    gemm_wmma_kernel<<<MT_PAD / 4, 256, 0, stream>>>((const v16bf*)Af, (const v16bf*)Wf,
                                                     support, N_NODES, K >> 5);
    zero_kernel<<<(unsigned)((n4 + 255) / 256), 256, 0, stream>>>((float4*)agg, n4);
    scatter_kernel<<<(unsigned)(((size_t)N_EDGES * 32 + 255) / 256), 256, 0, stream>>>(
        erow, ecol, eval, support, agg);
    finalize_kernel<<<(unsigned)(((size_t)MT_PAD * 16 * 64 + 255) / 256), 256, 0, stream>>>(
        agg, bias, out, Af);
  }
}
